// TPCLayer_26577257627926
// MI455X (gfx1250) — matrix operand order repaired and checked by reading, not compiled, not run on hardware
//
#include <hip/hip_runtime.h>

// ---------------------------------------------------------------------------
// TPC layer for MI455X (gfx1250, wave32).
// Dilated causal conv (implicit GEMM, K=64 per feature) + pointwise GEMM
// (K=1056) both on v_wmma_f32_16x16x32_bf16 with fp32 accumulate.
//
// Roofline: ~2.2 GFLOP vs ~235 MB mandatory traffic (201 MB fp32 output)
// -> needs >~215 TFLOP/s effective compute to stay HBM-bound, so matrix
// pipes are mandatory. All bf16 conversion + B-fragment swizzling is done
// ONCE in prep kernels (workspace), so the hot loops are pure
// global_load_b128 + wmma streams with zero LDS staging / zero cvt.
// ---------------------------------------------------------------------------

typedef __attribute__((ext_vector_type(16))) __bf16 v16bf;
typedef __attribute__((ext_vector_type(8)))  __bf16 v8bf;
typedef __attribute__((ext_vector_type(4)))  __bf16 v4bf;
typedef __attribute__((ext_vector_type(8)))  float  v8f;
typedef __attribute__((ext_vector_type(4)))  float  v4f;

#define B_  32
#define T_  256
#define F_  64
#define C_  16
#define CO_ 16
#define P_  64
#define S_  32
#define K_  4

#define SHUF16(lo, hi) __builtin_shufflevector(lo, hi, 0,1,2,3,4,5,6,7,8,9,10,11,12,13,14,15)

// workspace layout (bytes)
#define WS_PW   0                         // fp32 point_out   [8192][64]   2,097,152
#define WS_XB   2097152                   // bf16 x copy      [8192][1024] 16,777,216
#define WS_STB  18874368                  // bf16 static      [32][32]     2,048
#define WS_WPB  18876416                  // bf16 W_p swizzled (33 chunks) 135,168
#define WS_CWB  19011584                  // bf16 conv_w swizzled          131,072

// ===========================================================================
// Prep A: fp32 x -> bf16 xb (row-major, packed b64 stores)
// ===========================================================================
__global__ __launch_bounds__(256)
void prep_x(const float* __restrict__ x, __bf16* __restrict__ xb) {
  const long i = ((long)blockIdx.x * 256 + threadIdx.x) * 4;
  v4f v = *(const v4f*)(x + i);
  v4bf o = {(__bf16)v[0], (__bf16)v[1], (__bf16)v[2], (__bf16)v[3]};
  *(v4bf*)(xb + i) = o;
}

// ===========================================================================
// Prep B: static -> bf16; W_p and conv_w -> bf16 in WMMA B-fragment order.
// B-frag layout (32x16, 16-bit): lane n<16 holds col n K=0..7,16..23 (elem j: K=(j&7)+16*(j>>3)),
// lane n+16 holds col n K=8..15,24..31.  One element per thread; 524*256 threads.
// ===========================================================================
__global__ __launch_bounds__(256)
void prep_w(const float* __restrict__ stat, const float* __restrict__ Wp,
            const float* __restrict__ cw,
            __bf16* __restrict__ stb, __bf16* __restrict__ Wpb,
            __bf16* __restrict__ cwb) {
  const int idx = blockIdx.x * 256 + threadIdx.x;   // 0..134143
  if (idx < 1024) {                                  // static: 32x32
    stb[idx] = (__bf16)stat[idx];
  } else if (idx < 1024 + 67584) {                   // W_p: 33 chunks x 2048
    const int d    = idx - 1024;
    const int kc   = d >> 11;
    const int rem  = d & 2047;
    const int nt   = rem >> 9;
    const int lane = (rem >> 4) & 31;
    const int j    = rem & 15;
    const int col  = nt * 16 + (lane & 15);
    const int r    = (lane >> 4) * 8 + (j & 7) + ((j >> 3) << 4);  // K row within chunk
    Wpb[d] = (__bf16)Wp[(long)(kc * 32 + r) * P_ + col];
  } else {                                           // conv_w: 64 f x 2 q x 512
    const int d    = idx - (1024 + 67584);
    const int f    = d >> 10;
    const int rem  = d & 1023;
    const int q    = rem >> 9;
    const int n    = (rem >> 4) & 31;
    const int j    = rem & 15;                       // c = (n>>4)*8 + (j&7), tap = 2q + (j>>3)
    const int co   = n & 15;
    const int k    = 2 * q + (j >> 3);
    cwb[d] = (__bf16)cw[(((long)(f * CO_ + co)) * C_ + (n >> 4) * 8 + (j & 7)) * K_ + k];
  }
}

// ===========================================================================
// Kernel 1: point_out[B*T,64] = [x_flat | static] @ W_p + b_p  (fp32 -> pw)
// 256 threads = 8 independent waves; wave owns one 16x16 tile (mtile,ntile).
// Per K-chunk: 2x b128 A loads (xb) + 2x b128 B loads (Wpb) + 1 wmma.
// ===========================================================================
__global__ __launch_bounds__(256)
void tpc_pointwise_gemm(const __bf16* __restrict__ xb,
                        const __bf16* __restrict__ stb,
                        const __bf16* __restrict__ Wpb,
                        const float* __restrict__ bp,
                        float* __restrict__ pw) {
  const int tid  = threadIdx.x;
  const int lane = tid & 31;
  const int w    = blockIdx.x * 8 + (tid >> 5);   // 2048 waves
  const int mtile = w >> 2;                       // 512 M tiles of 16 rows
  const int nt    = w & 3;                        // 4 N tiles of 16 cols
  const int b  = mtile >> 4;
  const int t0 = (mtile & 15) << 4;
  const long rowbase = (long)(b * T_ + t0);

  const int arow = lane & 15;                     // A: M row this lane holds
  const int half = lane >> 4;                     // A: K-octet selector

  const __bf16* aptr = xb + (rowbase + arow) * (F_ * C_) + half * 8;
  const __bf16* bptr = Wpb + ((long)nt * 32 + lane) * 16;  // +2048/chunk

  v8f acc = {};
  for (int kc4 = 0; kc4 < 8; ++kc4) {
    __builtin_prefetch(bptr + 4 * 2048, 0, 1);    // next 4 B chunks (gfx1250)
#pragma unroll
    for (int u = 0; u < 4; ++u) {
      v8bf alo = *(const v8bf*)(aptr + u * 32);         // K k0+half*8..+7
      v8bf ahi = *(const v8bf*)(aptr + u * 32 + 16);    // K k0+16+half*8..
      v8bf blo = *(const v8bf*)(bptr + u * 2048);
      v8bf bhi = *(const v8bf*)(bptr + u * 2048 + 8);
      v16bf a  = SHUF16(alo, ahi);
      v16bf bf = SHUF16(blo, bhi);
      acc = __builtin_amdgcn_wmma_f32_16x16x32_bf16(
          false, a, false, bf, (short)0, acc, false, false);
    }
    aptr += 4 * 32;
    bptr += 4 * 2048;
  }
  // peeled chunk 32: static columns (row-independent within the M tile)
  {
    const __bf16* sp = stb + b * S_ + half * 8;
    v8bf alo = *(const v8bf*)sp;
    v8bf ahi = *(const v8bf*)(sp + 16);
    v8bf blo = *(const v8bf*)bptr;                // bptr now at chunk 32
    v8bf bhi = *(const v8bf*)(bptr + 8);
    v16bf a  = SHUF16(alo, ahi);
    v16bf bf = SHUF16(blo, bhi);
    acc = __builtin_amdgcn_wmma_f32_16x16x32_bf16(
        false, a, false, bf, (short)0, acc, false, false);
  }
  // epilogue: +bias, fp32 tile to pw.  C/D: lane n<16 -> col n rows 0..7;
  // lane n+16 -> col n rows 8..15.
  const int colp = nt * 16 + (lane & 15);
  const float bias = bp[colp];
  const int rsel = (lane >> 4) * 8;
#pragma unroll
  for (int v = 0; v < 8; ++v) {
    pw[(rowbase + rsel + v) * P_ + colp] = acc[v] + bias;
  }
}

// ===========================================================================
// Kernel 2: per-feature dilated causal conv (implicit GEMM, K=C*K=64) +
// fused concat([x, conv, point_bcast]) + ReLU writeback (the 201 MB stream).
// Block = 128 threads (4 waves) owns (f, 64 t rows); wave owns 16 rows.
// ===========================================================================
__global__ __launch_bounds__(128)
void tpc_conv_fuse(const float* __restrict__ x,
                   const __bf16* __restrict__ xb,
                   const __bf16* __restrict__ cwb,
                   const float* __restrict__ cb,
                   const float* __restrict__ pw,
                   float* __restrict__ out) {
  __shared__ float Cs[4][16][16];            // per-wave conv tile for writeback

  const int tid  = threadIdx.x;
  const int lane = tid & 31;
  const int wave = tid >> 5;
  const int f  = blockIdx.x >> 7;            // 64 features
  const int mt = blockIdx.x & 127;           // 128 m tiles of 64 rows
  const int b  = mt >> 2;
  const int t0 = ((mt & 3) << 6) + (wave << 4);
  const long rowbase = (long)(b * T_ + t0);

  const int arow  = lane & 15;
  const int half  = lane >> 4;
  const int t     = t0 + arow;
  const int cbase = f * C_ + half * 8;       // column base inside xb row

  v8f acc = {};
#pragma unroll
  for (int q = 0; q < 2; ++q) {
    // A frag elems 0..7: k=2q -> time t+4q-6; elems 8..15: k=2q+1 -> +2
    const int tt0 = t + 4 * q - 6;
    const int tt1 = tt0 + 2;
    v8bf alo = {}, ahi = {};
    if (tt0 >= 0)
      alo = *(const v8bf*)(xb + (long)(b * T_ + tt0) * (F_ * C_) + cbase);
    if (tt1 >= 0)
      ahi = *(const v8bf*)(xb + (long)(b * T_ + tt1) * (F_ * C_) + cbase);
    const __bf16* bbp = cwb + ((long)(f * 2 + q) * 32 + lane) * 16;
    v8bf blo = *(const v8bf*)bbp;
    v8bf bhi = *(const v8bf*)(bbp + 8);
    v16bf a  = SHUF16(alo, ahi);
    v16bf bf = SHUF16(blo, bhi);
    acc = __builtin_amdgcn_wmma_f32_16x16x32_bf16(
        false, a, false, bf, (short)0, acc, false, false);
  }

  // park conv tile (+bias) in LDS for the cooperative fused write
  const float bias = cb[f * CO_ + (lane & 15)];
#pragma unroll
  for (int v = 0; v < 8; ++v) {
    Cs[wave][(lane >> 4) * 8 + v][lane & 15] = acc[v] + bias;
  }
  __syncthreads();

  // fused concat + ReLU: 16 rows x 96 cols per wave, all b128 stores.
  // lane owns row r = lane&15; float4 segment seg = (lane>>4) + 2*p, p<12.
  const int r = lane & 15;
  const int halfseg = lane >> 4;
  const long grow = rowbase + r;
  float* orow = out + grow * (long)(F_ * 96) + (long)f * 96;
  const float* xrow = x  + (grow * F_ + f) * C_;   // exact fp32 passthrough
  const float* prow = pw + grow * P_;
#pragma unroll
  for (int p = 0; p < 12; ++p) {
    const int seg = halfseg + 2 * p;   // 0..23
    const int col = seg * 4;
    v4f v;
    if (col < 16)      v = *(const v4f*)(xrow + col);             // x
    else if (col < 32) v = *(const v4f*)&Cs[wave][r][col - 16];   // conv
    else               v = *(const v4f*)(prow + (col - 32));      // point
#pragma unroll
    for (int i = 0; i < 4; ++i) v[i] = fmaxf(v[i], 0.f);
    *(v4f*)(orow + col) = v;
  }
}

// ===========================================================================
extern "C" void kernel_launch(void* const* d_in, const int* in_sizes, int n_in,
                              void* d_out, int out_size, void* d_ws, size_t ws_size,
                              hipStream_t stream) {
  (void)in_sizes; (void)n_in; (void)out_size; (void)ws_size;
  const float* x  = (const float*)d_in[0];
  const float* st = (const float*)d_in[1];
  const float* cw = (const float*)d_in[2];
  const float* cb = (const float*)d_in[3];
  const float* Wp = (const float*)d_in[4];
  const float* bp = (const float*)d_in[5];
  // d_in[6] = dilation (==2), baked in.

  char* ws = (char*)d_ws;
  float*  pw  = (float*)(ws + WS_PW);
  __bf16* xbw = (__bf16*)(ws + WS_XB);
  __bf16* stb = (__bf16*)(ws + WS_STB);
  __bf16* Wpb = (__bf16*)(ws + WS_WPB);
  __bf16* cwb = (__bf16*)(ws + WS_CWB);

  prep_x<<<(B_ * T_ * F_ * C_) / 1024, 256, 0, stream>>>(x, xbw);
  prep_w<<<524, 256, 0, stream>>>(st, Wp, cw, stb, Wpb, cwb);
  tpc_pointwise_gemm<<<256, 256, 0, stream>>>(xbw, stb, Wpb, bp, pw);
  tpc_conv_fuse<<<F_ * (B_ * T_ / 64), 128, 0, stream>>>(x, xbw, cwb, cb, pw,
                                                         (float*)d_out);
}
